// MILSTM_Model_PN_45372034515223
// MI455X (gfx1250) — matrix operand order repaired and checked by reading, not compiled, not run on hardware
//
#include <hip/hip_runtime.h>

// ---------------------------------------------------------------------------
// MI455X (gfx1250) MI-LSTM implementation: bf16 WMMA (v_wmma_f32_16x16x32_bf16)
// for all recurrent matmuls, f32 accumulate, LDS-staged h/c state.
// Data movement: TDM tensor_load_to_lds for phase-2 activation tiles and
// global_store_async_from_lds for phase-1 state writeback (when available).
// ---------------------------------------------------------------------------

typedef __attribute__((ext_vector_type(16))) __bf16 v16bf;
typedef __attribute__((ext_vector_type(8)))  float  v8f;
typedef int v4i __attribute__((ext_vector_type(4)));

typedef __attribute__((address_space(1))) v4i g_v4i;   // global int4
typedef __attribute__((address_space(3))) v4i l_v4i;   // LDS int4

#if __has_builtin(__builtin_amdgcn_global_load_async_to_lds_b128) && \
    __has_builtin(__builtin_amdgcn_global_store_async_from_lds_b128)
#define HAVE_ASYNC_COPY 1
#else
#define HAVE_ASYNC_COPY 0
#endif

#if __has_builtin(__builtin_amdgcn_tensor_load_to_lds)
#define HAVE_TDM 1
#else
#define HAVE_TDM 0
#endif

constexpr int kH    = 64;
constexpr int kT    = 50;
constexpr int kB    = 2048;
constexpr int kNS   = 10;
constexpr int kNSTR = 21;                 // 1 + NS + NS streams
constexpr int kSEQS = kNSTR * kB;         // 43008 phase-1 sequences
constexpr int kDX   = kNSTR * kH;         // 1344 phase-2 input width

// packed phase-2 weight offsets (elements) inside Wp, each stored [N=64][K] bf16
constexpr size_t OFF_I0 = 0;              // K=128
constexpr size_t OFF_C0 = 8192;           // K=128
constexpr size_t OFF_F  = 16384;          // K=128
constexpr size_t OFF_O  = 24576;          // K=128
constexpr size_t OFF_I1 = 32768;          // K=704
constexpr size_t OFF_C1 = 77824;          // K=704
constexpr size_t OFF_I2 = 122880;         // K=704
constexpr size_t OFF_C2 = 167936;         // K=704
constexpr size_t OFF_A  = 212992;         // K=64
constexpr size_t WP_ELEMS = 217088;

__device__ __forceinline__ float fast_tanh(float x) {
#if __has_builtin(__builtin_amdgcn_tanhf)
  return __builtin_amdgcn_tanhf(x);       // v_tanh_f32 on gfx1250 (confirmed in asm)
#else
  return tanhf(x);
#endif
}
__device__ __forceinline__ float fast_sigmoid(float x) {
  return 0.5f * fast_tanh(0.5f * x) + 0.5f;
}
__device__ __forceinline__ __bf16 to_bf16(float x) { return (__bf16)x; }

__device__ __forceinline__ void wait_asynccnt0() {
#if __has_builtin(__builtin_amdgcn_s_wait_asynccnt)
  __builtin_amdgcn_s_wait_asynccnt(0);
#else
  asm volatile("s_wait_asynccnt 0x0" ::: "memory");
#endif
}

// Build AS-qualified typed pointers via integer round trips (LDS offset = low
// 32 bits of the generic LDS VA per the gfx1250 aperture rules).
__device__ __forceinline__ g_v4i* to_gptr4(const void* p) {
  return (g_v4i*)(size_t)(unsigned long long)p;
}
__device__ __forceinline__ l_v4i* to_lptr4(const void* p) {
  return (l_v4i*)(size_t)(unsigned)(size_t)p;
}

#if HAVE_TDM
typedef unsigned int u32x4 __attribute__((ext_vector_type(4)));
typedef int          i32x4 __attribute__((ext_vector_type(4)));
typedef int          i32x8 __attribute__((ext_vector_type(8)));

// One-line (1D) TDM copy: global -> LDS, data_size = 8B, tile_dim0 = bytes/8.
// D# layout per CDNA5 ISA 8.3/8.4: g0 = {flags, lds_addr, ga_lo, ga_hi|type2},
// g1 = {data_size, dims/tiles/strides}; groups 2/3 zero (<=2D tensor).
__device__ __forceinline__ void tdm_load_1d(const void* gsrc, const void* ldst,
                                            unsigned bytes) {
  const unsigned n  = bytes >> 3;                       // 8-byte elements
  const unsigned long long ga = (unsigned long long)gsrc;
  const unsigned lds = (unsigned)(size_t)ldst;          // LDS byte offset
  u32x4 g0;
  g0[0] = 1u;                                           // count=1, user mode
  g0[1] = lds;                                          // lds_addr
  g0[2] = (unsigned)ga;                                 // global_addr[31:0]
  g0[3] = (unsigned)(ga >> 32) | (2u << 30);            // global_addr[56:32] | type=2
  i32x8 g1;
  g1[0] = (int)(3u << 16);                              // data_size = 3 (8B)
  g1[1] = (int)((n & 0xFFFFu) << 16);                   // tensor_dim0[15:0]
  g1[2] = (int)((n >> 16) | (1u << 16));                // tensor_dim0[31:16], tensor_dim1=1
  g1[3] = (int)((n & 0xFFFFu) << 16);                   // tile_dim0
  g1[4] = 1;                                            // tile_dim1 = 1
  g1[5] = (int)n;                                       // tensor_dim0_stride[31:0]
  g1[6] = (int)((n & 0xFFFFu) << 16);                   // dim1_stride[15:0]
  g1[7] = 0;
  i32x4 z4 = {0, 0, 0, 0};
  i32x8 z8 = {0, 0, 0, 0, 0, 0, 0, 0};
  __builtin_amdgcn_tensor_load_to_lds(g0, g1, z4, z4, z8, 0);
}
#endif

union V16U { uint4 q[2]; v16bf v; };

// A fragment (16x32 bf16, MxK) from a row-major tile (rowstride rs elements).
// ISA layout: lanes 0-15 -> M=lane, K = kb+{0..7, 16..23}; lanes 16-31 -> K+8.
__device__ __forceinline__ v16bf load_a_frag(const __bf16* base, int rs, int kb) {
  const int lane = threadIdx.x & 31;
  const int hl   = lane >> 4;
  const int m    = lane & 15;
  const __bf16* p = base + m * rs + kb + hl * 8;
  V16U u;
  u.q[0] = *(const uint4*)(p);
  u.q[1] = *(const uint4*)(p + 16);
  return u.v;
}

// B fragment (32x16 bf16, KxN) from WT stored transposed as [N][K] row-major.
// ISA layout: lane's N = lane&15; lanes 0-15 hold K = kb+{0..7, 16..23}; lanes 16-31 -> K+8.
__device__ __forceinline__ v16bf load_b_frag(const __bf16* WT, int nb, int ktot, int kb) {
  const int lane = threadIdx.x & 31;
  const int hl   = lane >> 4;
  const __bf16* p = WT + (size_t)(nb + (lane & 15)) * ktot + kb + hl * 8;
  V16U u;
  u.q[0] = *(const uint4*)(p);
  u.q[1] = *(const uint4*)(p + 16);
  return u.v;
}

__device__ __forceinline__ v8f wmma_bf16(v16bf a, v16bf b, v8f c) {
  return __builtin_amdgcn_wmma_f32_16x16x32_bf16(false, a, false, b, (short)0, c,
                                                 false, false);
}

// ---------------------------------------------------------------------------
// Kernel 0a: gather series into time-major xpack[t][seq], seq = stream*2048 + b
// ---------------------------------------------------------------------------
__global__ __launch_bounds__(256)
void k_pack_x(const float* __restrict__ Y, const float* __restrict__ P,
              const float* __restrict__ Nn, float* __restrict__ xpack) {
  int idx = blockIdx.x * 256 + threadIdx.x;
  if (idx >= kT * kSEQS) return;
  int t = idx / kSEQS;
  int s = idx - t * kSEQS;
  int strm = s >> 11;
  int b    = s & (kB - 1);
  float v;
  if (strm == 0)        v = Y[(size_t)b * kT + t];
  else if (strm <= kNS) v = P[((size_t)b * kT + t) * kNS + (strm - 1)];
  else                  v = Nn[((size_t)b * kT + t) * kNS + (strm - 1 - kNS)];
  xpack[idx] = v;
}

// ---------------------------------------------------------------------------
// Kernel 0b: K1 hidden part -> transposed bf16 [N=256][K=64]
// ---------------------------------------------------------------------------
__global__ __launch_bounds__(256)
void k_conv_w1(const float* __restrict__ K1, __bf16* __restrict__ W1hT) {
  int idx = blockIdx.x * 256 + threadIdx.x;
  if (idx >= 256 * 64) return;
  int n = idx >> 6, k = idx & 63;
  W1hT[idx] = to_bf16(K1[(size_t)(1 + k) * 256 + n]);
}

// ---------------------------------------------------------------------------
// Kernel 0c: generic fp32 [K][64] -> bf16 transposed [64][K]
// ---------------------------------------------------------------------------
__global__ __launch_bounds__(256)
void k_transpose_bf16(const float* __restrict__ src, __bf16* __restrict__ dst, int K) {
  int idx = blockIdx.x * 256 + threadIdx.x;
  if (idx >= 64 * K) return;
  int n = idx / K, k = idx - n * K;
  dst[idx] = to_bf16(src[(size_t)k * 64 + n]);
}

// ---------------------------------------------------------------------------
// Kernel 1: 43008 independent scalar-input LSTMs, H=64, T=50.
// Block = 128 threads (4 waves); each wave owns 16 sequences.
// Per step per wave: z[16,256] = h[16,64] @ W1h  -> 32 bf16 WMMAs.
// c kept in registers in WMMA D-layout; h staged in LDS row-major for next A,
// streamed out with global_store_async_from_lds when available.
// ---------------------------------------------------------------------------
__global__ __launch_bounds__(128)
void k_phase1(const float* __restrict__ xpack, const __bf16* __restrict__ W1hT,
              const float* __restrict__ K1, const float* __restrict__ b1,
              __bf16* __restrict__ Xbuf) {
  __shared__ __bf16 sW[256 * 64];   // hidden weights, [N=256][K=64]
  __shared__ float  swx[256];       // input weight row (K1 row 0)
  __shared__ float  sb[256];        // bias
  __shared__ __bf16 sh[4][16 * 64]; // per-wave h tile, row-major
  __shared__ float  sx[4][16];      // per-wave current x

  const int tid  = threadIdx.x;
  const int wv   = tid >> 5;
  const int lane = tid & 31;
  const int hl   = lane >> 4;
  const int mcol = lane & 15;

  {
    const uint4* srcw = (const uint4*)W1hT;   // 32 KB = 2048 uint4
    uint4* dstw = (uint4*)sW;
    for (int i = tid; i < 2048; i += 128) dstw[i] = srcw[i];
    for (int i = tid; i < 256; i += 128) { swx[i] = K1[i]; sb[i] = b1[i]; }
  }
  {
    uint4 z = make_uint4(0u, 0u, 0u, 0u);
    uint4* dh = (uint4*)&sh[wv][0];
    for (int i = lane; i < 128; i += 32) dh[i] = z;
  }
  __syncthreads();

  const int seq0 = blockIdx.x * 64 + wv * 16;
  const int strm = seq0 >> 11;
  const int b0   = seq0 & (kB - 1);

  float creg[4][8];
#pragma unroll
  for (int g = 0; g < 4; ++g)
#pragma unroll
    for (int v = 0; v < 8; ++v) creg[g][v] = 0.f;

  for (int t = 0; t < kT; ++t) {
    if (lane < 16) sx[wv][lane] = xpack[(size_t)t * kSEQS + seq0 + lane];
    if (t + 1 < kT) __builtin_prefetch(xpack + (size_t)(t + 1) * kSEQS + seq0, 0, 1);

    v16bf a0 = load_a_frag(&sh[wv][0], kH, 0);
    v16bf a1 = load_a_frag(&sh[wv][0], kH, 32);

#if HAVE_ASYNC_COPY
    // previous step's async LDS->global stores must finish before sh rewrite
    wait_asynccnt0();
#endif

#pragma unroll
    for (int g = 0; g < 4; ++g) {
      v8f zi = {}, zj = {}, zf = {}, zo = {};
      zi = wmma_bf16(a0, load_b_frag(sW, g * 16,        kH, 0),  zi);
      zi = wmma_bf16(a1, load_b_frag(sW, g * 16,        kH, 32), zi);
      zj = wmma_bf16(a0, load_b_frag(sW, 64  + g * 16,  kH, 0),  zj);
      zj = wmma_bf16(a1, load_b_frag(sW, 64  + g * 16,  kH, 32), zj);
      zf = wmma_bf16(a0, load_b_frag(sW, 128 + g * 16,  kH, 0),  zf);
      zf = wmma_bf16(a1, load_b_frag(sW, 128 + g * 16,  kH, 32), zf);
      zo = wmma_bf16(a0, load_b_frag(sW, 192 + g * 16,  kH, 0),  zo);
      zo = wmma_bf16(a1, load_b_frag(sW, 192 + g * 16,  kH, 32), zo);
#pragma unroll
      for (int v = 0; v < 8; ++v) {
        const int row = v + 8 * hl;
        const int cH  = g * 16 + mcol;
        const float xr = sx[wv][row];
        const float vi = zi[v] + xr * swx[cH]        + sb[cH];
        const float vj = zj[v] + xr * swx[64  + cH]  + sb[64  + cH];
        const float vf = zf[v] + xr * swx[128 + cH]  + sb[128 + cH];
        const float vo = zo[v] + xr * swx[192 + cH]  + sb[192 + cH];
        const float cn = fast_sigmoid(vf + 1.0f) * creg[g][v] +
                         fast_sigmoid(vi) * fast_tanh(vj);
        const float hn = fast_sigmoid(vo) * fast_tanh(cn);
        creg[g][v] = cn;
        sh[wv][row * kH + cH] = to_bf16(hn);
      }
    }
    // stream full 16x64 h tile (bf16) to Xbuf[t][b][strm*64 ..]
    {
      const __bf16* srcp = &sh[wv][mcol * kH + hl * 32];
      __bf16* dstp = Xbuf + ((size_t)t * kB + b0 + mcol) * kDX + strm * kH + hl * 32;
#if HAVE_ASYNC_COPY
#pragma unroll
      for (int q = 0; q < 4; ++q)
        __builtin_amdgcn_global_store_async_from_lds_b128(
            to_gptr4(dstp + q * 8), to_lptr4(srcp + q * 8), 0, 0);
#else
      const uint4* srcv = (const uint4*)srcp;
      uint4* dstv = (uint4*)dstp;
      dstv[0] = srcv[0]; dstv[1] = srcv[1]; dstv[2] = srcv[2]; dstv[3] = srcv[3];
#endif
    }
  }
}

// ---------------------------------------------------------------------------
// Kernel 2: MI-LSTM with attention over the 3 branches. Block = 128 threads,
// 16 batch rows; wave w owns output col-tile w. ~106 WMMAs/wave/step.
// x tile staged to LDS via the Tensor Data Mover when available.
// ---------------------------------------------------------------------------
__global__ __launch_bounds__(128)
void k_phase2(const __bf16* __restrict__ Xbuf, const __bf16* __restrict__ Wp,
              const float* __restrict__ bi0, const float* __restrict__ bc0,
              const float* __restrict__ bi1, const float* __restrict__ bc1,
              const float* __restrict__ bi2, const float* __restrict__ bc2,
              const float* __restrict__ bfg, const float* __restrict__ bog,
              float* __restrict__ Hbuf) {
  __shared__ __bf16 sx[16 * kDX];   // x tile for this step (16 x 1344)
  __shared__ __bf16 shd[16 * kH];   // h (bf16 copy for A fragments)
  __shared__ __bf16 scd[16 * kH];   // c (bf16 copy for g = tanh(c@Wa))
  __shared__ float  su[48];         // u[row][3] attention logits

  const int tid   = threadIdx.x;
  const int wv    = tid >> 5;
  const int lane  = tid & 31;
  const int hl    = lane >> 4;
  const int mcol  = lane & 15;
  const int nb    = wv * 16;
  const int colg  = nb + mcol;
  const int brow0 = blockIdx.x * 16;

  for (int i = tid; i < 16 * kH; i += 128) { shd[i] = to_bf16(0.f); scd[i] = to_bf16(0.f); }

  float creg[8];
#pragma unroll
  for (int v = 0; v < 8; ++v) creg[v] = 0.f;

  const float c_bi0 = bi0[colg], c_bc0 = bc0[colg];
  const float c_bi1 = bi1[colg], c_bc1 = bc1[colg];
  const float c_bi2 = bi2[colg], c_bc2 = bc2[colg];
  const float c_bf  = bfg[colg], c_bo  = bog[colg];

  for (int t = 0; t < kT; ++t) {
    __syncthreads();
    if (tid < 48) su[tid] = 0.f;
#if HAVE_TDM
    if (wv == 0) {
      tdm_load_1d(Xbuf + ((size_t)t * kB + brow0) * kDX, sx, 16 * kDX * 2);
      __builtin_amdgcn_s_wait_tensorcnt(0);
    }
#elif HAVE_ASYNC_COPY
    {
      const uint4* src = (const uint4*)(Xbuf + ((size_t)t * kB + brow0) * kDX);
      uint4* dst = (uint4*)sx;
      for (int i = tid; i < (16 * kDX) / 8; i += 128)
        __builtin_amdgcn_global_load_async_to_lds_b128(
            to_gptr4(src + i), to_lptr4(dst + i), 0, 0);
      wait_asynccnt0();
    }
#else
    {
      const uint4* src = (const uint4*)(Xbuf + ((size_t)t * kB + brow0) * kDX);
      uint4* dst = (uint4*)sx;
      for (int i = tid; i < (16 * kDX) / 8; i += 128) dst[i] = src[i];
    }
#endif
    if (t + 1 < kT) __builtin_prefetch(Xbuf + ((size_t)(t + 1) * kB + brow0) * kDX, 0, 1);
    __syncthreads();

    v16bf ah0 = load_a_frag(shd, kH, 0);
    v16bf ah1 = load_a_frag(shd, kH, 32);

    float l0[8], l1[8], l2[8], gg[8], ff[8], oo[8];

    // branch 0 (xY + h, K=128) plus forget/output gates
    {
      v16bf ax0 = load_a_frag(sx, kDX, 0);
      v16bf ax1 = load_a_frag(sx, kDX, 32);
      v8f zi = {}, zc = {}, zf = {}, zo = {};
      zi = wmma_bf16(ax0, load_b_frag(Wp + OFF_I0, nb, 128, 0),  zi);
      zi = wmma_bf16(ax1, load_b_frag(Wp + OFF_I0, nb, 128, 32), zi);
      zi = wmma_bf16(ah0, load_b_frag(Wp + OFF_I0, nb, 128, 64), zi);
      zi = wmma_bf16(ah1, load_b_frag(Wp + OFF_I0, nb, 128, 96), zi);
      zc = wmma_bf16(ax0, load_b_frag(Wp + OFF_C0, nb, 128, 0),  zc);
      zc = wmma_bf16(ax1, load_b_frag(Wp + OFF_C0, nb, 128, 32), zc);
      zc = wmma_bf16(ah0, load_b_frag(Wp + OFF_C0, nb, 128, 64), zc);
      zc = wmma_bf16(ah1, load_b_frag(Wp + OFF_C0, nb, 128, 96), zc);
      zf = wmma_bf16(ax0, load_b_frag(Wp + OFF_F,  nb, 128, 0),  zf);
      zf = wmma_bf16(ax1, load_b_frag(Wp + OFF_F,  nb, 128, 32), zf);
      zf = wmma_bf16(ah0, load_b_frag(Wp + OFF_F,  nb, 128, 64), zf);
      zf = wmma_bf16(ah1, load_b_frag(Wp + OFF_F,  nb, 128, 96), zf);
      zo = wmma_bf16(ax0, load_b_frag(Wp + OFF_O,  nb, 128, 0),  zo);
      zo = wmma_bf16(ax1, load_b_frag(Wp + OFF_O,  nb, 128, 32), zo);
      zo = wmma_bf16(ah0, load_b_frag(Wp + OFF_O,  nb, 128, 64), zo);
      zo = wmma_bf16(ah1, load_b_frag(Wp + OFF_O,  nb, 128, 96), zo);
#pragma unroll
      for (int v = 0; v < 8; ++v) {
        l0[v] = fast_sigmoid(zi[v] + c_bi0) * fast_tanh(zc[v] + c_bc0);
        ff[v] = fast_sigmoid(zf[v] + c_bf);
        oo[v] = fast_sigmoid(zo[v] + c_bo);
      }
    }
    // branch 1 (xP + h, K=704)
    {
      v8f zi = {}, zc = {};
      for (int kb = 0; kb < 640; kb += 32) {
        v16bf a = load_a_frag(sx + 64, kDX, kb);
        zi = wmma_bf16(a, load_b_frag(Wp + OFF_I1, nb, 704, kb), zi);
        zc = wmma_bf16(a, load_b_frag(Wp + OFF_C1, nb, 704, kb), zc);
      }
      zi = wmma_bf16(ah0, load_b_frag(Wp + OFF_I1, nb, 704, 640), zi);
      zi = wmma_bf16(ah1, load_b_frag(Wp + OFF_I1, nb, 704, 672), zi);
      zc = wmma_bf16(ah0, load_b_frag(Wp + OFF_C1, nb, 704, 640), zc);
      zc = wmma_bf16(ah1, load_b_frag(Wp + OFF_C1, nb, 704, 672), zc);
#pragma unroll
      for (int v = 0; v < 8; ++v)
        l1[v] = fast_sigmoid(zi[v] + c_bi1) * fast_tanh(zc[v] + c_bc1);
    }
    // branch 2 (xN + h, K=704)
    {
      v8f zi = {}, zc = {};
      for (int kb = 0; kb < 640; kb += 32) {
        v16bf a = load_a_frag(sx + 704, kDX, kb);
        zi = wmma_bf16(a, load_b_frag(Wp + OFF_I2, nb, 704, kb), zi);
        zc = wmma_bf16(a, load_b_frag(Wp + OFF_C2, nb, 704, kb), zc);
      }
      zi = wmma_bf16(ah0, load_b_frag(Wp + OFF_I2, nb, 704, 640), zi);
      zi = wmma_bf16(ah1, load_b_frag(Wp + OFF_I2, nb, 704, 672), zi);
      zc = wmma_bf16(ah0, load_b_frag(Wp + OFF_C2, nb, 704, 640), zc);
      zc = wmma_bf16(ah1, load_b_frag(Wp + OFF_C2, nb, 704, 672), zc);
#pragma unroll
      for (int v = 0; v < 8; ++v)
        l2[v] = fast_sigmoid(zi[v] + c_bi2) * fast_tanh(zc[v] + c_bc2);
    }
    // g = tanh(c_prev @ Wa)
    {
      v8f zg = {};
      zg = wmma_bf16(load_a_frag(scd, kH, 0),  load_b_frag(Wp + OFF_A, nb, 64, 0),  zg);
      zg = wmma_bf16(load_a_frag(scd, kH, 32), load_b_frag(Wp + OFF_A, nb, 64, 32), zg);
#pragma unroll
      for (int v = 0; v < 8; ++v) gg[v] = fast_tanh(zg[v]);
    }

    // attention logits u[row][k] = sum_col l_k * g  (reduce cols: shfl + LDS atomics)
#pragma unroll
    for (int v = 0; v < 8; ++v) {
      float p0 = l0[v] * gg[v], p1 = l1[v] * gg[v], p2 = l2[v] * gg[v];
#pragma unroll
      for (int m = 1; m < 16; m <<= 1) {
        p0 += __shfl_xor(p0, m, 32);
        p1 += __shfl_xor(p1, m, 32);
        p2 += __shfl_xor(p2, m, 32);
      }
      if (mcol == 0) {
        const int row = v + 8 * hl;
        atomicAdd(&su[row * 3 + 0], p0);
        atomicAdd(&su[row * 3 + 1], p1);
        atomicAdd(&su[row * 3 + 2], p2);
      }
    }
    __syncthreads();

#pragma unroll
    for (int v = 0; v < 8; ++v) {
      const int row = v + 8 * hl;
      const float u0 = su[row * 3 + 0], u1 = su[row * 3 + 1], u2 = su[row * 3 + 2];
      const float mx = fmaxf(u0, fmaxf(u1, u2));
      const float e0 = __expf(u0 - mx), e1 = __expf(u1 - mx), e2 = __expf(u2 - mx);
      const float inv = 1.f / (e0 + e1 + e2);
      const float L  = (e0 * l0[v] + e1 * l1[v] + e2 * l2[v]) * inv;
      const float cn = ff[v] * creg[v] + L;
      const float hn = oo[v] * fast_tanh(cn);
      creg[v] = cn;
      shd[row * kH + colg] = to_bf16(hn);
      scd[row * kH + colg] = to_bf16(cn);
      Hbuf[((size_t)t * kB + brow0 + row) * kH + colg] = hn;
    }
  }
}

// ---------------------------------------------------------------------------
// Kernel 3: time attention + dense head. One wave per batch row.
// ---------------------------------------------------------------------------
__global__ __launch_bounds__(256)
void k_phase3(const float* __restrict__ Hbuf,
              const float* __restrict__ Wt,  const float* __restrict__ bt,
              const float* __restrict__ Wd1, const float* __restrict__ bd1,
              const float* __restrict__ Wd2, const float* __restrict__ bd2,
              float* __restrict__ out) {
  __shared__ float es[8][56];
  __shared__ float cs[8][64];
  const int tid = threadIdx.x, wv = tid >> 5, lane = tid & 31;
  const int b = blockIdx.x * 8 + wv;
  const float w0 = Wt[lane], w1 = Wt[lane + 32];

  for (int t = 0; t < kT; ++t) {
    const float* Hp = Hbuf + ((size_t)t * kB + b) * kH;
    float p = Hp[lane] * w0 + Hp[lane + 32] * w1;
#pragma unroll
    for (int m = 16; m >= 1; m >>= 1) p += __shfl_xor(p, m, 32);
    if (lane == 0) es[wv][t] = fast_tanh(p + bt[0]);
  }
  float mx = -3.4e38f;
  for (int t = 0; t < kT; ++t) mx = fmaxf(mx, es[wv][t]);
  float ssum = 0.f;
  for (int t = 0; t < kT; ++t) ssum += __expf(es[wv][t] - mx);
  const float inv = 1.f / ssum;

  float c0 = 0.f, c1 = 0.f;
  for (int t = 0; t < kT; ++t) {
    const float be = __expf(es[wv][t] - mx) * inv;
    const float* Hp = Hbuf + ((size_t)t * kB + b) * kH;
    c0 += be * Hp[lane];
    c1 += be * Hp[lane + 32];
  }
  cs[wv][lane] = c0; cs[wv][lane + 32] = c1;

  float a0 = bd1[lane], a1 = bd1[lane + 32];
  for (int k = 0; k < kH; ++k) {
    const float cv = cs[wv][k];
    a0 += cv * Wd1[(size_t)k * kH + lane];
    a1 += cv * Wd1[(size_t)k * kH + lane + 32];
  }
  const float r0 = fmaxf(a0, 0.f), r1 = fmaxf(a1, 0.f);
  float po = r0 * Wd2[lane] + r1 * Wd2[lane + 32];
#pragma unroll
  for (int m = 16; m >= 1; m >>= 1) po += __shfl_xor(po, m, 32);
  if (lane == 0) out[b] = po + bd2[0];
}

// ---------------------------------------------------------------------------
extern "C" void kernel_launch(void* const* d_in, const int* in_sizes, int n_in,
                              void* d_out, int out_size, void* d_ws, size_t ws_size,
                              hipStream_t stream) {
  (void)in_sizes; (void)n_in; (void)out_size; (void)ws_size;

  const float* Y   = (const float*)d_in[0];
  const float* P   = (const float*)d_in[1];
  const float* Nn  = (const float*)d_in[2];
  const float* K1  = (const float*)d_in[3];
  const float* b1  = (const float*)d_in[4];
  const float* Wc0 = (const float*)d_in[5];
  const float* bc0 = (const float*)d_in[6];
  const float* Wc1 = (const float*)d_in[7];
  const float* bc1 = (const float*)d_in[8];
  const float* Wc2 = (const float*)d_in[9];
  const float* bc2 = (const float*)d_in[10];
  const float* Wi0 = (const float*)d_in[11];
  const float* bi0 = (const float*)d_in[12];
  const float* Wi1 = (const float*)d_in[13];
  const float* bi1 = (const float*)d_in[14];
  const float* Wi2 = (const float*)d_in[15];
  const float* bi2 = (const float*)d_in[16];
  const float* Wf  = (const float*)d_in[17];
  const float* bfv = (const float*)d_in[18];
  const float* Wo  = (const float*)d_in[19];
  const float* bov = (const float*)d_in[20];
  const float* Wa  = (const float*)d_in[21];
  const float* Wt  = (const float*)d_in[22];
  const float* bt  = (const float*)d_in[23];
  const float* Wd1 = (const float*)d_in[24];
  const float* bd1 = (const float*)d_in[25];
  const float* Wd2 = (const float*)d_in[26];
  const float* bd2 = (const float*)d_in[27];

  char* ws = (char*)d_ws;
  size_t off = 0;
  auto alloc = [&](size_t bytes) -> char* {
    char* p = ws + off;
    off = (off + bytes + 255) & ~(size_t)255;
    return p;
  };
  float*  xpack = (float*)  alloc((size_t)kT * kSEQS * 4);        // 8.6 MB
  __bf16* W1hT  = (__bf16*) alloc((size_t)256 * 64 * 2);          // 32 KB
  __bf16* Xbuf  = (__bf16*) alloc((size_t)kT * kB * kDX * 2);     // 275 MB
  __bf16* Wp    = (__bf16*) alloc(WP_ELEMS * 2);                  // 434 KB
  float*  Hbuf  = (float*)  alloc((size_t)kT * kB * kH * 4);      // 26 MB

  k_pack_x<<<dim3((kT * kSEQS + 255) / 256), dim3(256), 0, stream>>>(Y, P, Nn, xpack);
  k_conv_w1<<<dim3((256 * 64 + 255) / 256), dim3(256), 0, stream>>>(K1, W1hT);

  auto tr = [&](const float* s, size_t offel, int K) {
    k_transpose_bf16<<<dim3((64 * K + 255) / 256), dim3(256), 0, stream>>>(s, Wp + offel, K);
  };
  tr(Wi0, OFF_I0, 128); tr(Wc0, OFF_C0, 128);
  tr(Wf,  OFF_F,  128); tr(Wo,  OFF_O,  128);
  tr(Wi1, OFF_I1, 704); tr(Wc1, OFF_C1, 704);
  tr(Wi2, OFF_I2, 704); tr(Wc2, OFF_C2, 704);
  tr(Wa,  OFF_A,  64);

  k_phase1<<<dim3(kSEQS / 64), dim3(128), 0, stream>>>(xpack, W1hT, K1, b1, Xbuf);
  k_phase2<<<dim3(kB / 16), dim3(128), 0, stream>>>(Xbuf, Wp, bi0, bc0, bi1, bc1,
                                                    bi2, bc2, bfv, bov, Hbuf);
  k_phase3<<<dim3(kB / 8), dim3(256), 0, stream>>>(Hbuf, Wt, bt, Wd1, bd1, Wd2, bd2,
                                                   (float*)d_out);
}
